// ComputeCorr_24017457119748
// MI455X (gfx1250) — matrix operand order repaired and checked by reading, not compiled, not run on hardware
//
#include <hip/hip_runtime.h>
#include <stdint.h>

typedef __attribute__((ext_vector_type(16))) __bf16 v16bf;
typedef __attribute__((ext_vector_type(8)))  float  v8f;

// Split fp32 into bf16 hi (truncated top 16 bits) + bf16 lo (rounded remainder).
// hi*hi + hi*lo + lo*hi reconstructs the product to ~2^-16 relative accuracy.
__device__ __forceinline__ void split2(float x, uint32_t& hi16, uint32_t& lo16) {
    union { float f; uint32_t u; } a; a.f = x;
    uint32_t hu = a.u & 0xFFFF0000u;
    union { uint32_t u; float f; } h; h.u = hu;
    float rem = x - h.f;
    __bf16 lb = (__bf16)rem;
    union { __bf16 b; uint16_t s; } lu; lu.b = lb;
    hi16 = hu >> 16;
    lo16 = (uint32_t)lu.s;
}

__device__ __forceinline__ __bf16 bits_to_bf16(uint32_t b) {
    union { uint16_t s; __bf16 b; } u; u.s = (uint16_t)b; return u.b;
}

// Flat generic pointer -> 32-bit LDS offset (LDS aperture uses addr[31:0]).
__device__ __forceinline__ uint32_t lds_off(const void* p) {
    return (uint32_t)(uintptr_t)p;
}

// Async global -> LDS copy, tracked by ASYNCcnt.
__device__ __forceinline__ void async_copy_b128(uint32_t dst_lds, const void* gaddr) {
    asm volatile("global_load_async_to_lds_b128 %0, %1, off"
                 :: "v"(dst_lds), "v"(gaddr) : "memory");
}
__device__ __forceinline__ void wait_async_le8() {
    asm volatile("s_wait_asynccnt 0x8" ::: "memory");
}
__device__ __forceinline__ void wait_async_0() {
    asm volatile("s_wait_asynccnt 0x0" ::: "memory");
}

// ---------- key-norm precompute: nn[b,n] = sum_d src_f[b,d,n]^2 ; kk likewise ----------
__global__ void norms_kernel(const float* __restrict__ src_f,
                             const float* __restrict__ tgt_f,
                             float* __restrict__ nn, float* __restrict__ kk,
                             int B, int D, int N, int M) {
    int idx = blockIdx.x * blockDim.x + threadIdx.x;
    int totalN = B * N;
    int totalM = B * M;
    if (idx < totalN) {
        int b = idx / N, n = idx % N;
        const float* p = src_f + (size_t)b * D * N + n;
        float s = 0.f;
        #pragma unroll 8
        for (int d = 0; d < D; ++d) { float v = p[(size_t)d * N]; s += v * v; }
        nn[idx] = s;
    } else if (idx < totalN + totalM) {
        int j = idx - totalN;
        int b = j / M, m = j % M;
        const float* p = tgt_f + (size_t)b * D * M + m;
        float s = 0.f;
        #pragma unroll 8
        for (int d = 0; d < D; ++d) { float v = p[(size_t)d * M]; s += v * v; }
        kk[j] = s;
    }
}

// ---------- fused flash-softmax correlation ----------
// out[b,n,:] = sum_m softmax_m(2*q_n.k_m - ||k_m||^2) * v[m,:]
// Keys are the WMMA A matrix, queries the B matrix, so C gives S^T[m,n]:
// lane = query column n (lane%16), VGPR j = key row m = 8*(lane/16)+j.
// Each lane-half runs an independent online softmax over its own key subset
// (no cross-lane ops in the hot loop); halves merge once at the end.
__global__ __launch_bounds__(128)
void corr_softmax_kernel(const float* __restrict__ qf,   // [B,D,Nq]
                         const float* __restrict__ kf,   // [B,D,Nk]
                         const float* __restrict__ knorm,// [B,Nk]
                         const float* __restrict__ vpts, // [B,Nk,3]
                         float* __restrict__ out,        // [B,Nq,3]
                         int D, int Nq, int Nk)
{
    __shared__ alignas(16) float  sRaw[2][64 * 64];      // raw f32 key chunk (double buffer, async dst)
    __shared__ alignas(32) uint4  sAhi[4 * 2 * 32 * 2];  // key A-fragments hi  (8 KB)
    __shared__ alignas(32) uint4  sAlo[4 * 2 * 32 * 2];  // key A-fragments lo  (8 KB)
    __shared__ alignas(16) float4 sKV[64];               // {vx, vy, vz, ||k||^2} per key column

    const int tid  = threadIdx.x;
    const int lane = tid & 31;
    const int g    = lane >> 4;      // lane half -> which 8 key rows of each tile
    const int mc   = lane & 15;      // query column within wave tile
    const int b    = blockIdx.y;
    const int nbase = blockIdx.x * 64 + (tid >> 5) * 16;

    const float* qfb = qf    + (size_t)b * D * Nq;
    const float* kfb = kf    + (size_t)b * D * Nk;
    const float* knb = knorm + (size_t)b * Nk;
    const float* vb  = vpts  + (size_t)b * Nk * 3;

    // ---- persistent query B fragments (16-bit 32x16 layout: K = 8g + h (h<8), 16 + 8g + (h-8) (h>=8), col = lane%16)
    v16bf q_hi[2], q_lo[2];
    {
        const int ncol = nbase + mc;
        #pragma unroll
        for (int s = 0; s < 2; ++s) {
            #pragma unroll
            for (int h = 0; h < 16; ++h) {
                int d = s * 32 + 8 * g + h + (h >= 8 ? 8 : 0);
                uint32_t hb, lb;
                split2(qfb[(size_t)d * Nq + ncol], hb, lb);
                q_hi[s][h] = bits_to_bf16(hb);
                q_lo[s][h] = bits_to_bf16(lb);
            }
        }
    }

    // per-lane online softmax state for ONE query column over this half's keys
    float m_run = -3.0e38f, lsum = 0.f, ax = 0.f, ay = 0.f, az = 0.f;

    uint32_t* pAhi = (uint32_t*)sAhi;
    uint32_t* pAlo = (uint32_t*)sAlo;

    const int nchunks = Nk >> 6;

    // prologue: async-load chunk 0 (64 cols x 64 d of f32) into buffer 0
    #pragma unroll
    for (int i = 0; i < 8; ++i) {
        int u  = i * 128 + tid;          // 16-byte units: 16 per d-row
        int d  = u >> 4;
        int c4 = (u & 15) * 4;
        async_copy_b128(lds_off(&sRaw[0][d * 64 + c4]),
                        kfb + (size_t)d * Nk + c4);
    }

    for (int ch = 0; ch < nchunks; ++ch) {
        const int mbase = ch << 6;
        const int buf   = ch & 1;

        // issue async for next chunk into the other buffer, then wait for current
        if (ch + 1 < nchunks) {
            #pragma unroll
            for (int i = 0; i < 8; ++i) {
                int u  = i * 128 + tid;
                int d  = u >> 4;
                int c4 = (u & 15) * 4;
                async_copy_b128(lds_off(&sRaw[buf ^ 1][d * 64 + c4]),
                                kfb + (size_t)d * Nk + (mbase + 64) + c4);
            }
            wait_async_le8();            // in-order: first 8 (current chunk) done
        } else {
            wait_async_0();
        }
        __syncthreads();                 // all waves' async data landed; prev compute done

        // ---- convert raw f32 -> key A-fragments (16-bit 16x32 A layout), hi/lo split.
        // element (d,m): K=d%32; K<16 -> laneGroup=K/8,h=K%8 ; K>=16 -> laneGroup=(K-16)/8,h=8+(K-16)%8
        for (int it = 0; it < 16; ++it) {
            int w = it * 128 + tid;      // 2048 items: 32 d-pairs x 64 cols
            int mlocal = w & 63;
            int d0 = (w >> 6) * 2;
            float f0 = sRaw[buf][d0 * 64 + mlocal];
            float f1 = sRaw[buf][(d0 + 1) * 64 + mlocal];
            uint32_t h0, l0, h1, l1;
            split2(f0, h0, l0);
            split2(f1, h1, l1);
            int t = mlocal >> 4;
            int r = mlocal & 15;         // A row = key m within tile = lane%16
            int s  = d0 >> 5;
            int kl = d0 & 31;
            int gg, h;
            if (kl < 16) { gg = kl >> 3;        h = kl & 7; }
            else         { gg = (kl - 16) >> 3; h = 8 + ((kl - 16) & 7); }
            int laneA = gg * 16 + r;
            int widx  = ((t * 2 + s) * 32 + laneA) * 8 + (h >> 1);
            pAhi[widx] = h0 | (h1 << 16);
            pAlo[widx] = l0 | (l1 << 16);
        }
        if (tid < 64) {
            int m = mbase + tid;
            sKV[tid] = make_float4(vb[(size_t)m * 3 + 0], vb[(size_t)m * 3 + 1],
                                   vb[(size_t)m * 3 + 2], knb[m]);
        }
        __syncthreads();

        // ---- compute: 4 key tiles of 16 rows ----
        #pragma unroll
        for (int t = 0; t < 4; ++t) {
            v16bf ah0 = *(const v16bf*)&sAhi[((t * 2 + 0) * 32 + lane) * 2];
            v16bf ah1 = *(const v16bf*)&sAhi[((t * 2 + 1) * 32 + lane) * 2];
            v16bf al0 = *(const v16bf*)&sAlo[((t * 2 + 0) * 32 + lane) * 2];
            v16bf al1 = *(const v16bf*)&sAlo[((t * 2 + 1) * 32 + lane) * 2];

            v8f c = {};
            // bf16x3: hi*hi + hi*lo + lo*hi over two K=32 slices (D=64)
            c = __builtin_amdgcn_wmma_f32_16x16x32_bf16(false, ah0, false, q_hi[0], (short)0, c, false, false);
            c = __builtin_amdgcn_wmma_f32_16x16x32_bf16(false, ah1, false, q_hi[1], (short)0, c, false, false);
            c = __builtin_amdgcn_wmma_f32_16x16x32_bf16(false, ah0, false, q_lo[0], (short)0, c, false, false);
            c = __builtin_amdgcn_wmma_f32_16x16x32_bf16(false, ah1, false, q_lo[1], (short)0, c, false, false);
            c = __builtin_amdgcn_wmma_f32_16x16x32_bf16(false, al0, false, q_hi[0], (short)0, c, false, false);
            c = __builtin_amdgcn_wmma_f32_16x16x32_bf16(false, al1, false, q_hi[1], (short)0, c, false, false);

            // this lane's 8 key rows: m = tile*16 + 8g + j (contiguous)
            float4 kv[8];
            float  z[8];
            #pragma unroll
            for (int j = 0; j < 8; ++j) {
                kv[j] = sKV[t * 16 + 8 * g + j];
                z[j]  = fmaf(2.0f, c[j], -kv[j].w);   // logit = 2*q.k - ||k||^2
            }
            // within-lane tile max (tree)
            float a01 = fmaxf(z[0], z[1]), a23 = fmaxf(z[2], z[3]);
            float a45 = fmaxf(z[4], z[5]), a67 = fmaxf(z[6], z[7]);
            float tmax = fmaxf(fmaxf(a01, a23), fmaxf(a45, a67));

            float mnew  = fmaxf(m_run, tmax);
            float scale = __expf(m_run - mnew);
            m_run = mnew;

            float p[8];
            #pragma unroll
            for (int j = 0; j < 8; ++j) p[j] = __expf(z[j] - mnew);

            float ps = ((p[0] + p[1]) + (p[2] + p[3])) + ((p[4] + p[5]) + (p[6] + p[7]));
            float px = 0.f, py = 0.f, pz = 0.f;
            #pragma unroll
            for (int j = 0; j < 8; ++j) {
                px = fmaf(p[j], kv[j].x, px);
                py = fmaf(p[j], kv[j].y, py);
                pz = fmaf(p[j], kv[j].z, pz);
            }
            lsum = fmaf(lsum, scale, ps);
            ax   = fmaf(ax,   scale, px);
            ay   = fmaf(ay,   scale, py);
            az   = fmaf(az,   scale, pz);
        }
    }

    // ---- merge the two independent lane-half softmaxes (keys split by half) ----
    float mo  = __shfl_xor(m_run, 16, 32);
    float lo2 = __shfl_xor(lsum,  16, 32);
    float axo = __shfl_xor(ax,    16, 32);
    float ayo = __shfl_xor(ay,    16, 32);
    float azo = __shfl_xor(az,    16, 32);
    float Mt  = fmaxf(m_run, mo);
    float se  = __expf(m_run - Mt);
    float so  = __expf(mo - Mt);
    float lt  = lsum * se + lo2 * so;
    float inv = 1.0f / lt;
    if (g == 0) {
        int n = nbase + mc;
        float* o = out + ((size_t)b * Nq + n) * 3;
        o[0] = (ax * se + axo * so) * inv;
        o[1] = (ay * se + ayo * so) * inv;
        o[2] = (az * se + azo * so) * inv;
    }
}

extern "C" void kernel_launch(void* const* d_in, const int* in_sizes, int n_in,
                              void* d_out, int out_size, void* d_ws, size_t ws_size,
                              hipStream_t stream) {
    const float* src   = (const float*)d_in[0];   // [B,N,3]
    const float* tgt   = (const float*)d_in[1];   // [B,M,3]
    const float* src_f = (const float*)d_in[2];   // [B,D,N]
    const float* tgt_f = (const float*)d_in[3];   // [B,D,M]
    float* out = (float*)d_out;

    const int B = 4, N = 4096, M = 4096, D = 64;

    float* nn = (float*)d_ws;                 // [B,N] src-feature norms
    float* kk = nn + (size_t)B * N;           // [B,M] tgt-feature norms

    int tot = B * (N + M);
    norms_kernel<<<(tot + 255) / 256, 256, 0, stream>>>(src_f, tgt_f, nn, kk, B, D, N, M);

    // src_corr: queries = src features, keys = tgt features, values = tgt points
    dim3 grid1(N / 64, B);
    corr_softmax_kernel<<<grid1, 128, 0, stream>>>(src_f, tgt_f, kk, tgt, out, D, N, M);

    // tgt_corr: queries = tgt features, keys = src features, values = src points
    dim3 grid2(M / 64, B);
    corr_softmax_kernel<<<grid2, 128, 0, stream>>>(tgt_f, src_f, nn, src,
                                                   out + (size_t)B * N * 3, D, M, N);
}
